// LFLSTM_8151847927867
// MI455X (gfx1250) — matrix operand order repaired and checked by reading, not compiled, not run on hardware
//
#include <hip/hip_runtime.h>

typedef __attribute__((ext_vector_type(16))) __bf16 v16bf;
typedef __attribute__((ext_vector_type(8)))  __bf16 bf16x8;
typedef __attribute__((ext_vector_type(8)))  float  v8f;

#define T_STEPS 128
#define SUMH    384

static __device__ __forceinline__ __bf16 f2bf(float f) {
    union { float f; unsigned u; } in; in.f = f;
    unsigned u = in.u;
    u += 0x7FFFu + ((u >> 16) & 1u);          // round-to-nearest-even
    unsigned short s = (unsigned short)(u >> 16);
    union { unsigned short s; __bf16 b; } out; out.s = s;
    return out.b;
}

static __device__ __forceinline__ float sigm(float x) {
    return 1.f / (1.f + __expf(-x));
}
static __device__ __forceinline__ float tanh_fast(float x) {
    return 1.f - 2.f / (__expf(2.f * x) + 1.f);
}

// One workgroup = 16 batch rows, runs all T steps of one LSTM.
// A-matrix in LDS: [16][Kpad] bf16 = [x_t (Dp cols, zero-padded) | h (H cols)].
// Weights: Wcat [4H][Kpad] bf16 = [W_ih (padded) | W_hh], rows contiguous in K.
// x_{t+1} is double-buffered through registers so its global loads overlap the
// WMMA burst of step t; x_{t+2} is prefetched (global_prefetch_b8).
template<int H, int D, int Dp>
static __device__ void lstm_block(const float* __restrict__ x,      // [B,T,D]
                                  const __bf16* __restrict__ Wcat,  // [4H][Kpad]
                                  const float* __restrict__ bsum,   // [4H] = b_ih+b_hh
                                  float* __restrict__ lasth,        // [B,SUMH]
                                  int col_off, int bb, __bf16* As)
{
    constexpr int Kpad = Dp + H;
    constexpr int KT   = Kpad / 32;      // k-tiles of 32
    constexpr int JT   = H / 16;         // hidden-column tiles
    constexpr int MAXJ = (JT + 7) / 8;   // j-tiles per wave (8 waves)
    constexpr int NSTAGE = (16 * Dp) / 256;  // x elements per thread (exact)

    const int tid  = threadIdx.x;
    const int wave = tid >> 5;
    const int lane = tid & 31;
    const int half = lane >> 4;
    const int ln16 = lane & 15;

    // zero h region of A
    for (int idx = tid; idx < 16 * H; idx += 256) {
        int r = idx / H, j = idx - r * H;
        As[r * Kpad + Dp + j] = f2bf(0.f);
    }

    // hoist time-invariant gate biases into registers (per wave / j-tile)
    float bias[MAXJ][4];
    {
        int ji = 0;
        for (int jt = wave; jt < JT; jt += 8, ++ji) {
#pragma unroll
            for (int q = 0; q < 4; ++q)
                bias[ji][q] = bsum[q * H + jt * 16 + ln16];
        }
    }

    v8f cfrag[MAXJ];
#pragma unroll
    for (int i = 0; i < MAXJ; ++i) cfrag[i] = (v8f)0.f;

    // prologue: load x_0 into registers
    float xreg[NSTAGE];
#pragma unroll
    for (int s = 0; s < NSTAGE; ++s) {
        int idx = tid + s * 256;
        int r = idx / Dp, d = idx - r * Dp;
        xreg[s] = (d < D) ? x[(size_t)(bb + r) * (T_STEPS * D) + d] : 0.f;
    }

    for (int t = 0; t < T_STEPS; ++t) {
        // commit staged x_t (fp32 -> bf16) into A[:, 0:Dp)
#pragma unroll
        for (int s = 0; s < NSTAGE; ++s) {
            int idx = tid + s * 256;
            int r = idx / Dp, d = idx - r * Dp;
            As[r * Kpad + d] = f2bf(xreg[s]);
        }
        __syncthreads();   // x staged + h from previous step visible

        // kick off x_{t+1} register loads: they overlap the WMMA burst below
        if (t + 1 < T_STEPS) {
#pragma unroll
            for (int s = 0; s < NSTAGE; ++s) {
                int idx = tid + s * 256;
                int r = idx / Dp, d = idx - r * Dp;
                xreg[s] = (d < D)
                    ? x[(size_t)(bb + r) * (T_STEPS * D) + (size_t)(t + 1) * D + d]
                    : 0.f;
            }
        }
        // prefetch x_{t+2} into cache (global_prefetch_b8), 128B stride
        if (t + 2 < T_STEPS) {
            for (int e = tid * 32; e < 16 * Dp; e += 256 * 32) {
                int r = e / Dp, d = e - r * Dp;
                if (d < D)
                    __builtin_prefetch(
                        &x[(size_t)(bb + r) * (T_STEPS * D) + (size_t)(t + 2) * D + d],
                        0, 1);
            }
        }

        // gates = [x_t | h] @ Wcat^T  via v_wmma_f32_16x16x32_bf16 (bias added later)
        v8f acc[MAXJ][4];
        {
            int ji = 0;
            for (int jt = wave; jt < JT; jt += 8, ++ji) {
                const int jbase = jt * 16;
#pragma unroll
                for (int q = 0; q < 4; ++q)
                    acc[ji][q] = (v8f)0.f;

                for (int kt = 0; kt < KT; ++kt) {
                    const int k0 = kt * 32;
                    // A fragment (16x32, from LDS): lane = row m, halves split K
                    union { v16bf v; bf16x8 h[2]; } ua;
                    const __bf16* arow = As + ln16 * Kpad + k0;
                    ua.h[0] = *(const bf16x8*)(arow + half * 8);
                    ua.h[1] = *(const bf16x8*)(arow + 16 + half * 8);
#pragma unroll
                    for (int q = 0; q < 4; ++q) {
                        // B fragment (32x16): lane = col n, halves split K like A
                        union { v16bf v; bf16x8 h[2]; } ub;
                        const __bf16* brow =
                            Wcat + (size_t)(q * H + jbase + ln16) * Kpad + k0 + half * 8;
                        ub.h[0] = *(const bf16x8*)(brow);
                        ub.h[1] = *(const bf16x8*)(brow + 16);
                        acc[ji][q] = __builtin_amdgcn_wmma_f32_16x16x32_bf16(
                            false, ua.v, false, ub.v, (short)0, acc[ji][q], false, false);
                    }
                }
            }
        }
        __syncthreads();   // all reads of A done before rewriting h region

        // LSTM cell update in registers; write new h (bf16) into A[:, Dp:Dp+H)
        {
            int ji = 0;
            for (int jt = wave; jt < JT; jt += 8, ++ji) {
                const int jcol = jt * 16 + ln16;
                v8f c = cfrag[ji];
#pragma unroll
                for (int r = 0; r < 8; ++r) {
                    float iv = sigm(acc[ji][0][r] + bias[ji][0]);
                    float fv = sigm(acc[ji][1][r] + bias[ji][1]);
                    float gv = tanh_fast(acc[ji][2][r] + bias[ji][2]);
                    float ov = sigm(acc[ji][3][r] + bias[ji][3]);
                    float cv = fv * c[r] + iv * gv;
                    float hv = ov * tanh_fast(cv);
                    c[r] = cv;
                    const int mrow = r + half * 8;
                    As[mrow * Kpad + Dp + jcol] = f2bf(hv);
                    if (t == T_STEPS - 1)
                        lasth[(size_t)(bb + mrow) * SUMH + col_off + jcol] = hv;
                }
                cfrag[ji] = c;
            }
        }
        __syncthreads();   // h writes done before next step's x commit
    }
}

__global__ __launch_bounds__(256, 1) void lstm_fused_kernel(
    const float* x0, const float* x1, const float* x2,
    const __bf16* W0, const __bf16* W1, const __bf16* W2,
    const float* bs0, const float* bs1, const float* bs2,
    float* lasth)
{
    __shared__ __attribute__((aligned(16))) __bf16 As[16 * 576];  // max Kpad=576 -> 18KB
    const int blk = blockIdx.x;
    if (blk < 32)       lstm_block<256, 300, 320>(x0, W0, bs0, lasth, 0,   blk * 16,        As);
    else if (blk < 64)  lstm_block<64,  74,  96 >(x1, W1, bs1, lasth, 256, (blk - 32) * 16, As);
    else                lstm_block<64,  35,  64 >(x2, W2, bs2, lasth, 320, (blk - 64) * 16, As);
}

// Pack [W_ih | W_hh] -> bf16 Wcat[4H][Dp+H] (zero-padded K) and bsum = b_ih+b_hh.
__global__ void build_wcat_kernel(const float* __restrict__ w_ih, const float* __restrict__ w_hh,
                                  const float* __restrict__ b_ih, const float* __restrict__ b_hh,
                                  __bf16* __restrict__ Wcat, float* __restrict__ bsum,
                                  int G, int D, int Dp, int H)
{
    int idx = blockIdx.x * blockDim.x + threadIdx.x;
    int Kpad = Dp + H;
    int total = G * Kpad;
    if (idx < total) {
        int g = idx / Kpad, k = idx - g * Kpad;
        float v = (k < Dp) ? ((k < D) ? w_ih[g * D + k] : 0.f)
                           : w_hh[g * H + (k - Dp)];
        Wcat[idx] = f2bf(v);
    }
    if (idx < G) bsum[idx] = b_ih[idx] + b_hh[idx];
}

// out[b] = sum_j relu(lasth[b,:] . w1[j,:] + b1[j]) * w2[j] + b2   (tiny, fp32 VALU)
__global__ __launch_bounds__(64) void head_kernel(const float* __restrict__ lasth,
                                                  const float* __restrict__ w1,
                                                  const float* __restrict__ b1,
                                                  const float* __restrict__ w2,
                                                  const float* __restrict__ b2,
                                                  float* __restrict__ out)
{
    __shared__ float sh[64];
    const int b = blockIdx.x, j = threadIdx.x;
    const float* row = lasth + (size_t)b * SUMH;
    const float* wr  = w1 + (size_t)j * SUMH;
    float acc = b1[j];
    for (int k = 0; k < SUMH; ++k) acc += row[k] * wr[k];
    sh[j] = fmaxf(acc, 0.f) * w2[j];
    __syncthreads();
    if (j == 0) {
        float s = b2[0];
        for (int k = 0; k < 64; ++k) s += sh[k];
        out[b] = s;
    }
}

extern "C" void kernel_launch(void* const* d_in, const int* in_sizes, int n_in,
                              void* d_out, int out_size, void* d_ws, size_t ws_size,
                              hipStream_t stream) {
    (void)in_sizes; (void)n_in; (void)out_size; (void)ws_size;

    const float* x0 = (const float*)d_in[0];
    const float* x1 = (const float*)d_in[1];
    const float* x2 = (const float*)d_in[2];
    const float* w_ih0 = (const float*)d_in[3];
    const float* w_hh0 = (const float*)d_in[4];
    const float* b_ih0 = (const float*)d_in[5];
    const float* b_hh0 = (const float*)d_in[6];
    const float* w_ih1 = (const float*)d_in[7];
    const float* w_hh1 = (const float*)d_in[8];
    const float* b_ih1 = (const float*)d_in[9];
    const float* b_hh1 = (const float*)d_in[10];
    const float* w_ih2 = (const float*)d_in[11];
    const float* w_hh2 = (const float*)d_in[12];
    const float* b_ih2 = (const float*)d_in[13];
    const float* b_hh2 = (const float*)d_in[14];
    const float* w1 = (const float*)d_in[15];
    const float* b1 = (const float*)d_in[16];
    const float* w2 = (const float*)d_in[17];
    const float* b2 = (const float*)d_in[18];

    char* ws = (char*)d_ws;
    size_t off = 0;
    auto alloc = [&](size_t bytes) -> void* {
        off = (off + 255) & ~(size_t)255;
        void* p = ws + off;
        off += bytes;
        return p;
    };

    __bf16* W0  = (__bf16*)alloc((size_t)1024 * 576 * 2);
    __bf16* W1c = (__bf16*)alloc((size_t)256 * 160 * 2);
    __bf16* W2c = (__bf16*)alloc((size_t)256 * 128 * 2);
    float*  bs0 = (float*)alloc(1024 * 4);
    float*  bs1 = (float*)alloc(256 * 4);
    float*  bs2 = (float*)alloc(256 * 4);
    float*  lasth = (float*)alloc((size_t)512 * SUMH * 4);

    build_wcat_kernel<<<(1024 * 576 + 255) / 256, 256, 0, stream>>>(
        w_ih0, w_hh0, b_ih0, b_hh0, W0, bs0, 1024, 300, 320, 256);
    build_wcat_kernel<<<(256 * 160 + 255) / 256, 256, 0, stream>>>(
        w_ih1, w_hh1, b_ih1, b_hh1, W1c, bs1, 256, 74, 96, 64);
    build_wcat_kernel<<<(256 * 128 + 255) / 256, 256, 0, stream>>>(
        w_ih2, w_hh2, b_ih2, b_hh2, W2c, bs2, 256, 35, 64, 64);

    lstm_fused_kernel<<<96, 256, 0, stream>>>(x0, x1, x2, W0, W1c, W2c, bs0, bs1, bs2, lasth);

    head_kernel<<<512, 64, 0, stream>>>(lasth, w1, b1, w2, b2, (float*)d_out);
}
